// ISTFTProcessor_48576080118714
// MI455X (gfx1250) — matrix operand order repaired and checked by reading, not compiled, not run on hardware
//
#include <hip/hip_runtime.h>
#include <hip/hip_bf16.h>

#define N_FFT   4096
#define HOP     1024
#define N_FREQ  2049
#define FRAMES  512
#define BC      8
#define OUT_LEN 523264   // (FRAMES-1)*HOP
#define PAD     2048     // N_FFT/2, center=True trim

#define TM 128           // C macro-tile rows (t)
#define TN 128           // C macro-tile cols (n)
#define TK 32            // K chunk (bf16 WMMA depth)
#define KFULL 64         // full 32-wide chunks per basis (64*32 = 2048)
#define ASTR 72          // LDS row stride in bf16 (64 data + 8 pad; 144B, 16B-aligned)

typedef __attribute__((ext_vector_type(16))) __bf16 v16bf;
typedef __attribute__((ext_vector_type(8)))  float  v8f;

union V16U { int4 q[2]; v16bf v; };

__device__ __forceinline__ void split_bf16(float x, __bf16& h, __bf16& l) {
    h = (__bf16)x;
    l = (__bf16)(x - (float)h);
}

// ---------------------------------------------------------------------------
// Kernel 1: fused dual-GEMM (cos*real + sin*imag) with bf16 hi/lo split WMMA,
// f32 accumulation. K = 2*2048 in unguarded 32-chunks; the leftover Nyquist
// column (f = 2048) is folded into the epilogue as exact fp32 FMAs. Window is
// applied in the epilogue; windowed frames written to xws[bc][t][n].
// ---------------------------------------------------------------------------
__global__ __launch_bounds__(256)
void istft_gemm_kernel(const float* __restrict__ realp,
                       const float* __restrict__ imagp,
                       const float* __restrict__ cosb,
                       const float* __restrict__ sinb,
                       const float* __restrict__ win,
                       float* __restrict__ xws) {
    __shared__ __bf16 ldsA[TM][ASTR];  // row t: [0,32)=hi(k), [32,64)=lo(k)
    __shared__ __bf16 ldsB[TN][ASTR];  // col n: [0,32)=hi(k), [32,64)=lo(k)

    const int tid   = threadIdx.x;
    const int tileM = blockIdx.x * TM;
    const int tileN = blockIdx.y * TN;
    const int bc    = blockIdx.z;

    const int lane = tid & 31;
    const int m16  = lane & 15;
    const int sel  = lane >> 4;       // lane half selector
    const int wm   = (tid >> 5) & 3;  // wave M quadrant (4 x 32 rows)
    const int wn   = (tid >> 7) & 1;  // wave N half     (2 x 64 cols)

    v8f acc[2][4];
    #pragma unroll
    for (int i = 0; i < 2; ++i)
        #pragma unroll
        for (int j = 0; j < 4; ++j) acc[i][j] = {};

    for (int it = 0; it < 2 * KFULL; ++it) {
        const int    phase = it >> 6;           // 0: cos/real, 1: sin/imag
        const int    kc    = it & (KFULL - 1);
        const int    f0    = kc * TK;
        const float* basis = phase ? sinb  : cosb;
        const float* spec  = (phase ? imagp : realp) + (size_t)bc * N_FREQ * FRAMES;

        __syncthreads();

        // ---- stage A (basis tile: TM rows of t, TK freqs) -> hi/lo, no guards
        {
            const int c4 = (tid & 7) * 4;   // 0..28
            const int r0 = tid >> 3;        // 0..31
            #pragma unroll
            for (int rr = 0; rr < 4; ++rr) {
                const int row = r0 + rr * 32;
                const float* src = basis + (size_t)(tileM + row) * N_FREQ + f0 + c4;
                __builtin_prefetch(src + TK, 0, 3);
                float vals[4];
                #pragma unroll
                for (int j = 0; j < 4; ++j) vals[j] = src[j];
                #pragma unroll
                for (int j = 0; j < 4; ++j) {
                    __bf16 h, l; split_bf16(vals[j], h, l);
                    ldsA[row][c4 + j]      = h;
                    ldsA[row][TK + c4 + j] = l;
                }
            }
        }
        // ---- stage B (spec tile: TK freqs x TN frames) -> transpose + hi/lo
        {
            const int n4 = (tid & 31) * 4;  // 0..124
            const int fr = tid >> 5;        // 0..7
            #pragma unroll
            for (int ff = 0; ff < 4; ++ff) {
                const int fl = fr + ff * 8;
                const float4 v = *(const float4*)(spec + (size_t)(f0 + fl) * FRAMES
                                                  + tileN + n4);
                const float vv[4] = {v.x, v.y, v.z, v.w};
                #pragma unroll
                for (int j = 0; j < 4; ++j) {
                    __bf16 h, l; split_bf16(vv[j], h, l);
                    ldsB[n4 + j][fl]      = h;
                    ldsB[n4 + j][TK + fl] = l;
                }
            }
        }

        __syncthreads();

        // ---- load WMMA fragments in ISA wave32 layouts
        // A 16x32 bf16: lanes 0-15 K{0..7,16..23}, lanes 16-31 K{8..15,24..31}
        V16U aHi[2], aLo[2];
        #pragma unroll
        for (int sm = 0; sm < 2; ++sm) {
            const __bf16* pr = &ldsA[wm * 32 + sm * 16 + m16][0];
            aHi[sm].q[0] = *(const int4*)(pr + sel * 8);
            aHi[sm].q[1] = *(const int4*)(pr + 16 + sel * 8);
            aLo[sm].q[0] = *(const int4*)(pr + TK + sel * 8);
            aLo[sm].q[1] = *(const int4*)(pr + TK + 16 + sel * 8);
        }
        // B 32x16 bf16: lanes 0-15 K{0..7,16..23}, lanes 16-31 K{8..15,24..31}, N=lane%16
        V16U bHi[4], bLo[4];
        #pragma unroll
        for (int sn = 0; sn < 4; ++sn) {
            const __bf16* pc = &ldsB[wn * 64 + sn * 16 + m16][0];
            bHi[sn].q[0] = *(const int4*)(pc + sel * 8);
            bHi[sn].q[1] = *(const int4*)(pc + 16 + sel * 8);
            bLo[sn].q[0] = *(const int4*)(pc + TK + sel * 8);
            bLo[sn].q[1] = *(const int4*)(pc + TK + 16 + sel * 8);
        }

        // ---- 2x4 subtiles x 3 terms (hi*hi + hi*lo + lo*hi) of bf16 split
        #pragma unroll
        for (int sm = 0; sm < 2; ++sm) {
            #pragma unroll
            for (int sn = 0; sn < 4; ++sn) {
                acc[sm][sn] = __builtin_amdgcn_wmma_f32_16x16x32_bf16(
                    false, aHi[sm].v, false, bHi[sn].v, (short)0, acc[sm][sn], false, false);
                acc[sm][sn] = __builtin_amdgcn_wmma_f32_16x16x32_bf16(
                    false, aHi[sm].v, false, bLo[sn].v, (short)0, acc[sm][sn], false, false);
                acc[sm][sn] = __builtin_amdgcn_wmma_f32_16x16x32_bf16(
                    false, aLo[sm].v, false, bHi[sn].v, (short)0, acc[sm][sn], false, false);
            }
        }
    }

    // ---- K tail: f = 2048 (Nyquist), exact fp32 FMA into accumulators
    // C layout: VGPR j -> M = j + sel*8, lane%16 -> N
    const float* realN = realp + ((size_t)bc * N_FREQ + 2048) * FRAMES;
    const float* imagN = imagp + ((size_t)bc * N_FREQ + 2048) * FRAMES;
    float rN[4], iN[4];
    #pragma unroll
    for (int sn = 0; sn < 4; ++sn) {
        const int n = tileN + wn * 64 + sn * 16 + m16;
        rN[sn] = realN[n];
        iN[sn] = imagN[n];
    }
    #pragma unroll
    for (int sm = 0; sm < 2; ++sm) {
        const int tBase = tileM + wm * 32 + sm * 16 + sel * 8;
        #pragma unroll
        for (int j = 0; j < 8; ++j) {
            const float cN = cosb[(size_t)(tBase + j) * N_FREQ + 2048];
            const float sN = sinb[(size_t)(tBase + j) * N_FREQ + 2048];
            #pragma unroll
            for (int sn = 0; sn < 4; ++sn)
                acc[sm][sn][j] += cN * rN[sn] + sN * iN[sn];
        }
    }

    // ---- epilogue: apply window, store windowed frames to workspace
    #pragma unroll
    for (int sm = 0; sm < 2; ++sm) {
        const int tBase = tileM + wm * 32 + sm * 16 + sel * 8;
        float wv[8];
        #pragma unroll
        for (int j = 0; j < 8; ++j) wv[j] = win[tBase + j];
        #pragma unroll
        for (int sn = 0; sn < 4; ++sn) {
            const int n = tileN + wn * 64 + sn * 16 + m16;
            float* dst = xws + ((size_t)bc * N_FFT + tBase) * FRAMES + n;
            #pragma unroll
            for (int j = 0; j < 8; ++j)
                dst[(size_t)j * FRAMES] = acc[sm][sn][j] * wv[j];
        }
    }
}

// ---------------------------------------------------------------------------
// Kernel 2: overlap-add + window-energy normalization + center trim.
// One thread per output sample; <=4 overlapping frames (HOP = N_FFT/4).
// ---------------------------------------------------------------------------
__global__ __launch_bounds__(256)
void istft_ola_kernel(const float* __restrict__ xws,
                      const float* __restrict__ win,
                      float* __restrict__ out) {
    const int bc = blockIdx.y;
    const int so = blockIdx.x * blockDim.x + threadIdx.x;
    if (so >= OUT_LEN) return;

    const int s = so + PAD;
    int nLo = (s - (N_FFT - 1) + (HOP - 1)) / HOP;  // ceil((s-4095)/1024)
    if (nLo < 0) nLo = 0;
    int nHi = s / HOP;
    if (nHi > FRAMES - 1) nHi = FRAMES - 1;

    const float* xb = xws + (size_t)bc * N_FFT * FRAMES;
    float acc = 0.0f, wsum = 0.0f;
    for (int n = nLo; n <= nHi; ++n) {
        const int t = s - n * HOP;
        const float w = win[t];
        acc  += xb[(size_t)t * FRAMES + n];  // already multiplied by win[t]
        wsum += w * w;
    }
    out[(size_t)bc * OUT_LEN + so] = acc / fmaxf(wsum, 1e-8f);
}

extern "C" void kernel_launch(void* const* d_in, const int* in_sizes, int n_in,
                              void* d_out, int out_size, void* d_ws, size_t ws_size,
                              hipStream_t stream) {
    const float* realp = (const float*)d_in[0];
    const float* imagp = (const float*)d_in[1];
    // d_in[2] = length scalar (fixed 523264, compile-time constant here)
    const float* cosb  = (const float*)d_in[3];
    const float* sinb  = (const float*)d_in[4];
    const float* win   = (const float*)d_in[5];

    float* xws = (float*)d_ws;           // needs BC*N_FFT*FRAMES*4 = 64 MiB
    float* out = (float*)d_out;

    dim3 g1(N_FFT / TM, FRAMES / TN, BC);   // (32, 4, 8)
    istft_gemm_kernel<<<g1, 256, 0, stream>>>(realp, imagp, cosb, sinb, win, xws);

    dim3 g2((OUT_LEN + 255) / 256, BC);
    istft_ola_kernel<<<g2, 256, 0, stream>>>(xws, win, out);
}
